// RefGatedMLPFusedMoE_44049184588304
// MI455X (gfx1250) — matrix operand order repaired and checked by reading, not compiled, not run on hardware
//
#include <hip/hip_runtime.h>
#include <hip/hip_bf16.h>
#include <cstdint>
#include <cstddef>

typedef __bf16 bf16;
typedef __attribute__((ext_vector_type(16))) __bf16 v16bf;
typedef __attribute__((ext_vector_type(8)))  __bf16 v8bf;
typedef __attribute__((ext_vector_type(8)))  float  v8f;

static constexpr int T_TOK = 4096;   // tokens
static constexpr int HDIM  = 2048;   // hidden
static constexpr int IDIM  = 4096;   // intermediate
static constexpr int NEXP  = 8;      // experts
static constexpr int BLK_M = 128;    // token-tile
static constexpr int BLK_N = 64;     // out-feature tile
static constexpr int BK    = 32;     // K step == one WMMA K
static constexpr int LDS_P = BK + 8; // 40 bf16 = 80B row stride: bank-conflict free, 16B aligned
static constexpr int NTHR  = 256;    // 8 waves (wave32)

union Frag { v16bf v; v8bf h[2]; };

__device__ __forceinline__ v8f zero_v8f() {
  v8f z;
#pragma unroll
  for (int i = 0; i < 8; ++i) z[i] = 0.0f;
  return z;
}

__device__ __forceinline__ v8bf cvt8(float4 a, float4 b) {
  v8bf r;
  r[0] = (bf16)a.x; r[1] = (bf16)a.y; r[2] = (bf16)a.z; r[3] = (bf16)a.w;
  r[4] = (bf16)b.x; r[5] = (bf16)b.y; r[6] = (bf16)b.z; r[7] = (bf16)b.w;
  return r;
}

__device__ __forceinline__ v8f wmma_bf16(const Frag& a, const Frag& b, v8f c) {
  // D(f32 16x16) = A(bf16 16x32) * B(bf16 32x16) + C
  return __builtin_amdgcn_wmma_f32_16x16x32_bf16(
      /*neg_a=*/false, a.v, /*neg_b=*/false, b.v,
      /*c_mod=*/(short)0, c, /*reuse_a=*/false, /*reuse_b=*/false);
}

// Async copy of 16 bytes global -> LDS (CDNA5 GLOBAL_LOAD_ASYNC_TO_LDS_B128,
// tracked by ASYNCcnt; bypasses VGPRs). For AMDGPU flat pointers into the LDS
// aperture, addr[31:0] is the wave-relative LDS byte offset, which is exactly
// what the VDST operand expects.
__device__ __forceinline__ void async_stage16(const bf16* __restrict__ g, bf16* l) {
#if defined(__gfx1250__)
  unsigned lds_off = (unsigned)(size_t)l;
  asm volatile("global_load_async_to_lds_b128 %0, %1, off"
               :: "v"(lds_off), "v"(g) : "memory");
#else
  *(v8bf*)l = *(const v8bf*)g;
#endif
}

__device__ __forceinline__ void wait_async0() {
#if defined(__gfx1250__)
  asm volatile("s_wait_asynccnt 0x0" ::: "memory");
#endif
}

// ---------------------------------------------------------------------------
// Kernel 0: zero output + expert counters
// ---------------------------------------------------------------------------
__global__ void moe_zero_kernel(float4* __restrict__ out4, int n4, int* __restrict__ cnt) {
  int i = blockIdx.x * blockDim.x + threadIdx.x;
  if (i < n4) out4[i] = make_float4(0.f, 0.f, 0.f, 0.f);
  if (i < NEXP) cnt[i] = 0;
}

// ---------------------------------------------------------------------------
// Kernel 1: top-2 router + softmax; build per-expert token/weight lists
// ---------------------------------------------------------------------------
__global__ void moe_router_kernel(const float* __restrict__ logits,
                                  int* __restrict__ cnt,
                                  int* __restrict__ tokl,
                                  float* __restrict__ wtl) {
  int t = blockIdx.x * blockDim.x + threadIdx.x;
  if (t >= T_TOK) return;
  float v[NEXP];
#pragma unroll
  for (int e = 0; e < NEXP; ++e) v[e] = logits[t * NEXP + e];
  int e0 = 0; float v0 = v[0];
#pragma unroll
  for (int e = 1; e < NEXP; ++e) if (v[e] > v0) { v0 = v[e]; e0 = e; }
  int e1 = -1; float v1 = -3.4e38f;
#pragma unroll
  for (int e = 0; e < NEXP; ++e) if (e != e0 && v[e] > v1) { v1 = v[e]; e1 = e; }
  // softmax over the two selected logits (v0 >= v1)
  float w0 = 1.0f / (1.0f + __expf(v1 - v0));
  float w1 = 1.0f - w0;
  int s0 = atomicAdd(&cnt[e0], 1);
  tokl[e0 * T_TOK + s0] = t; wtl[e0 * T_TOK + s0] = w0;
  int s1 = atomicAdd(&cnt[e1], 1);
  tokl[e1 * T_TOK + s1] = t; wtl[e1 * T_TOK + s1] = w1;
}

// ---------------------------------------------------------------------------
// Kernel 2: exclusive prefix sum over 8 expert counts
// ---------------------------------------------------------------------------
__global__ void moe_offsets_kernel(const int* __restrict__ cnt, int* __restrict__ offs) {
  if (threadIdx.x == 0) {
    int a = 0;
#pragma unroll
    for (int e = 0; e < NEXP; ++e) { offs[e] = a; a += cnt[e]; }
  }
}

// ---------------------------------------------------------------------------
// Kernel 3: per-expert gate/up GEMM + silu*up fused, act stored bf16.
//   gate = Xg @ w1[e]^T ; up = Xg @ w3[e]^T ; act = silu(gate)*up
//   A rows gathered by token list; both operands K-contiguous (NT GEMM).
//   fp32 -> bf16 conversion must pass through VGPRs, so this kernel uses
//   register staging (no async path possible for converting streams).
// ---------------------------------------------------------------------------
__global__ __launch_bounds__(NTHR)
void moe_gateup_kernel(const float* __restrict__ X,
                       const float* __restrict__ W1,
                       const float* __restrict__ W3,
                       const int*   __restrict__ tokl,
                       const int*   __restrict__ cnt,
                       const int*   __restrict__ offs,
                       bf16*        __restrict__ act) {
  const int e     = blockIdx.z;
  const int m0    = blockIdx.y * BLK_M;
  const int nbase = blockIdx.x * BLK_N;
  const int n_e   = cnt[e];
  if (m0 >= n_e) return;            // block-uniform early exit
  const int obase = offs[e];

  __shared__ int sTok[BLK_M];
  __shared__ alignas(16) bf16 sA [BLK_M][LDS_P];
  __shared__ alignas(16) bf16 sB1[BLK_N][LDS_P];
  __shared__ alignas(16) bf16 sB3[BLK_N][LDS_P];

  const int tid = threadIdx.x;
  if (tid < BLK_M) {
    int m = m0 + tid;
    sTok[tid] = (m < n_e) ? tokl[e * T_TOK + m] : 0;  // clamp: safe dummy row
  }
  __syncthreads();

  // loader coordinates
  const int ar = tid & (BLK_M - 1), ap = tid >> 7;   // 128 rows x 2 halves (16 f32)
  const int br = tid & (BLK_N - 1), bp = tid >> 6;   // 64 rows x 4 parts   (8 f32)
  // wave/lane coordinates (wave32)
  const int wave = tid >> 5, lane = tid & 31;
  const int lh = lane & 15, lg = lane >> 4;
  const int wm = wave & 3, wn = wave >> 2;           // 4x2 waves -> 128x64 tile

  v8f accG[2][2], accU[2][2];
#pragma unroll
  for (int i = 0; i < 2; ++i)
#pragma unroll
    for (int j = 0; j < 2; ++j) { accG[i][j] = zero_v8f(); accU[i][j] = zero_v8f(); }

  const int tokA = sTok[ar];
  const float* __restrict__ srcA  = X  + (size_t)tokA * HDIM + ap * 16;
  const float* __restrict__ srcB1 = W1 + ((size_t)e * IDIM + nbase + br) * HDIM + bp * 8;
  const float* __restrict__ srcB3 = W3 + ((size_t)e * IDIM + nbase + br) * HDIM + bp * 8;

  for (int kk = 0; kk < HDIM; kk += BK) {
    { // stage A (gathered token rows), fp32 -> bf16
      const float4* p = (const float4*)(srcA + kk);
      float4 f0 = p[0], f1 = p[1], f2 = p[2], f3 = p[3];
      *(v8bf*)&sA[ar][ap * 16]     = cvt8(f0, f1);
      *(v8bf*)&sA[ar][ap * 16 + 8] = cvt8(f2, f3);
    }
    { // stage w1 / w3 tiles, fp32 -> bf16
      const float4* p1 = (const float4*)(srcB1 + kk);
      float4 g0 = p1[0], g1 = p1[1];
      *(v8bf*)&sB1[br][bp * 8] = cvt8(g0, g1);
      const float4* p3 = (const float4*)(srcB3 + kk);
      float4 u0 = p3[0], u1 = p3[1];
      *(v8bf*)&sB3[br][bp * 8] = cvt8(u0, u1);
      if (kk + BK < HDIM) {
        __builtin_prefetch(srcB1 + kk + BK, 0, 0);
        __builtin_prefetch(srcB3 + kk + BK, 0, 0);
      }
    }
    __syncthreads();

    Frag af[2], b1f[2], b3f[2];
#pragma unroll
    for (int am = 0; am < 2; ++am) {   // A: contiguous k = 16*lg + j per lane
      const int rr = wm * 32 + am * 16 + lh;
      af[am].h[0] = *(const v8bf*)&sA[rr][16 * lg];
      af[am].h[1] = *(const v8bf*)&sA[rr][16 * lg + 8];
    }
#pragma unroll
    for (int bn = 0; bn < 2; ++bn) {   // B: contiguous k = 16*lg + j per lane
      const int cc = wn * 32 + bn * 16 + lh;
      b1f[bn].h[0] = *(const v8bf*)&sB1[cc][16 * lg];
      b1f[bn].h[1] = *(const v8bf*)&sB1[cc][16 * lg + 8];
      b3f[bn].h[0] = *(const v8bf*)&sB3[cc][16 * lg];
      b3f[bn].h[1] = *(const v8bf*)&sB3[cc][16 * lg + 8];
    }
#pragma unroll
    for (int am = 0; am < 2; ++am)
#pragma unroll
      for (int bn = 0; bn < 2; ++bn) {
        accG[am][bn] = wmma_bf16(af[am], b1f[bn], accG[am][bn]);
        accU[am][bn] = wmma_bf16(af[am], b3f[bn], accU[am][bn]);
      }
    __syncthreads();
  }

  // epilogue: act = silu(gate) * up  (bf16)
#pragma unroll
  for (int am = 0; am < 2; ++am)
#pragma unroll
    for (int bn = 0; bn < 2; ++bn) {
      const int col = nbase + wn * 32 + bn * 16 + lh;
#pragma unroll
      for (int i = 0; i < 8; ++i) {           // D: m = i + 8*lg, n = lh
        const int s = m0 + wm * 32 + am * 16 + i + 8 * lg;
        if (s < n_e) {
          float g = accG[am][bn][i];
          float u = accU[am][bn][i];
          float h = (g / (1.0f + __expf(-g))) * u;
          act[(size_t)(obase + s) * IDIM + col] = (bf16)h;
        }
      }
    }
}

// ---------------------------------------------------------------------------
// Kernel 4: per-expert down GEMM: y = act @ w2[e]^T, scaled scatter-add.
//   A operand (act) is already bf16 -> staged with GLOBAL_LOAD_ASYNC_TO_LDS
//   (ASYNCcnt, no VGPR round-trip). B (w2, fp32) converts through VGPRs.
//   Deterministic: each token gets exactly 2 atomic adds onto zeroed out.
// ---------------------------------------------------------------------------
__global__ __launch_bounds__(NTHR)
void moe_down_kernel(const bf16*  __restrict__ act,
                     const float* __restrict__ W2,
                     const int*   __restrict__ tokl,
                     const float* __restrict__ wtl,
                     const int*   __restrict__ cnt,
                     const int*   __restrict__ offs,
                     float*       __restrict__ out) {
  const int e     = blockIdx.z;
  const int m0    = blockIdx.y * BLK_M;
  const int nbase = blockIdx.x * BLK_N;   // column base in H
  const int n_e   = cnt[e];
  if (m0 >= n_e) return;
  const int obase = offs[e];

  __shared__ int   sTok[BLK_M];
  __shared__ float sWt [BLK_M];
  __shared__ alignas(16) bf16 sA[BLK_M][LDS_P];
  __shared__ alignas(16) bf16 sB[BLK_N][LDS_P];

  const int tid = threadIdx.x;
  if (tid < BLK_M) {
    int m = m0 + tid;
    if (m < n_e) { sTok[tid] = tokl[e * T_TOK + m]; sWt[tid] = wtl[e * T_TOK + m]; }
    else         { sTok[tid] = -1;                  sWt[tid] = 0.0f; }
  }
  __syncthreads();

  const int ar = tid & (BLK_M - 1), ap = tid >> 7;
  const int br = tid & (BLK_N - 1), bp = tid >> 6;
  const int wave = tid >> 5, lane = tid & 31;
  const int lh = lane & 15, lg = lane >> 4;
  const int wm = wave & 3, wn = wave >> 2;

  v8f acc[2][2];
#pragma unroll
  for (int i = 0; i < 2; ++i)
#pragma unroll
    for (int j = 0; j < 2; ++j) acc[i][j] = zero_v8f();

  const int mrA = (m0 + ar < n_e) ? (m0 + ar) : (n_e - 1);   // clamp OOB rows
  const bf16*  __restrict__ srcA = act + (size_t)(obase + mrA) * IDIM + ap * 16;
  const float* __restrict__ srcB = W2 + ((size_t)e * HDIM + nbase + br) * IDIM + bp * 8;
  bf16* ldsA0 = &sA[ar][ap * 16];
  bf16* ldsA1 = &sA[ar][ap * 16 + 8];

  for (int kk = 0; kk < IDIM; kk += BK) {
    // A (bf16): async DMA global -> LDS, no VGPR staging
    async_stage16(srcA + kk,     ldsA0);
    async_stage16(srcA + kk + 8, ldsA1);
    { // stage w2 tile fp32 -> bf16 through VGPRs
      const float4* p = (const float4*)(srcB + kk);
      float4 g0 = p[0], g1 = p[1];
      *(v8bf*)&sB[br][bp * 8] = cvt8(g0, g1);
    }
    if (kk + BK < IDIM) __builtin_prefetch(srcB + kk + BK, 0, 0);
    wait_async0();        // this wave's async->LDS complete
    __syncthreads();      // all waves' LDS writes visible

    Frag af[2], bf_[2];
#pragma unroll
    for (int am = 0; am < 2; ++am) {
      const int rr = wm * 32 + am * 16 + lh;
      af[am].h[0] = *(const v8bf*)&sA[rr][16 * lg];
      af[am].h[1] = *(const v8bf*)&sA[rr][16 * lg + 8];
    }
#pragma unroll
    for (int bn = 0; bn < 2; ++bn) {
      const int cc = wn * 32 + bn * 16 + lh;
      bf_[bn].h[0] = *(const v8bf*)&sB[cc][16 * lg];
      bf_[bn].h[1] = *(const v8bf*)&sB[cc][16 * lg + 8];
    }
#pragma unroll
    for (int am = 0; am < 2; ++am)
#pragma unroll
      for (int bn = 0; bn < 2; ++bn)
        acc[am][bn] = wmma_bf16(af[am], bf_[bn], acc[am][bn]);
    __syncthreads();
  }

  // scatter: out[tok, col] += wt * y
#pragma unroll
  for (int am = 0; am < 2; ++am)
#pragma unroll
    for (int bn = 0; bn < 2; ++bn) {
      const int col = nbase + wn * 32 + bn * 16 + lh;
#pragma unroll
      for (int i = 0; i < 8; ++i) {
        const int ml = wm * 32 + am * 16 + i + 8 * lg;  // local row 0..127
        const int t  = sTok[ml];
        if (t >= 0)
          atomicAdd(&out[(size_t)t * HDIM + col], sWt[ml] * acc[am][bn][i]);
      }
    }
}

// ---------------------------------------------------------------------------
// Host launcher
// ---------------------------------------------------------------------------
extern "C" void kernel_launch(void* const* d_in, const int* in_sizes, int n_in,
                              void* d_out, int out_size, void* d_ws, size_t ws_size,
                              hipStream_t stream) {
  (void)in_sizes; (void)n_in; (void)out_size; (void)ws_size;
  const float* X      = (const float*)d_in[0];  // [T, H]
  const float* logits = (const float*)d_in[1];  // [T, E]
  const float* W1     = (const float*)d_in[2];  // [E, I, H]
  const float* W3     = (const float*)d_in[3];  // [E, I, H]
  const float* W2     = (const float*)d_in[4];  // [E, H, I]
  float* out = (float*)d_out;                   // [T, H]

  // workspace layout
  char*  ws   = (char*)d_ws;
  int*   cnt  = (int*)(ws + 0);                                   // 8 ints
  int*   offs = (int*)(ws + 256);                                 // 8 ints
  int*   tokl = (int*)(ws + 512);                                 // [E][T]
  float* wtl  = (float*)(ws + 512 + (size_t)NEXP * T_TOK * 4);    // [E][T]
  bf16*  act  = (bf16*)(ws + 512 + (size_t)NEXP * T_TOK * 8);     // [2T][I] bf16 (64 MB)

  const int n4 = T_TOK * HDIM / 4;
  moe_zero_kernel<<<(n4 + 255) / 256, 256, 0, stream>>>((float4*)out, n4, cnt);
  moe_router_kernel<<<(T_TOK + 255) / 256, 256, 0, stream>>>(logits, cnt, tokl, wtl);
  moe_offsets_kernel<<<1, 32, 0, stream>>>(cnt, offs);

  dim3 g1(IDIM / BLK_N, T_TOK / BLK_M, NEXP);   // (64, 32, 8), worst-case M, early exit
  moe_gateup_kernel<<<g1, NTHR, 0, stream>>>(X, W1, W3, tokl, cnt, offs, act);

  dim3 g2(HDIM / BLK_N, T_TOK / BLK_M, NEXP);   // (32, 32, 8)
  moe_down_kernel<<<g2, NTHR, 0, stream>>>(act, W2, tokl, wtl, cnt, offs, out);
}
